// Cbow_11244224381331
// MI455X (gfx1250) — matrix operand order repaired and checked by reading, not compiled, not run on hardware
//
#include <hip/hip_runtime.h>
#include <hip/hip_bf16.h>
#include <math.h>

// CBOW negative-sampling loss on MI455X (gfx1250, wave32).
// B=16384, C=20, N=20, D=128. Memory-bound gather; both tables (102 MB) are
// L2-resident (192 MB). Scores for 16 b-rows computed at once as the diagonal
// of v_wmma_f32_16x16x32_f16 tiles; f32 accumulate, f16 inputs (values ~1e-2,
// well inside f16 normal range).

typedef __attribute__((ext_vector_type(16))) _Float16 v16h;
typedef __attribute__((ext_vector_type(8)))  float    v8f;
typedef __attribute__((ext_vector_type(2)))  _Float16 h2;

#define EMB   128
#define CWIN  20
#define NNEG  20
#define BTOT  16384
#define TILE  16            // b-rows per wave (one WMMA tile edge)
#define WAVES 4             // waves per block
#define PITCH 132           // halves per LDS row (128 + 4 pad -> conflict-free)

// Wave-private LDS phases: same-wave DS ops are in-order (DScnt), so only a
// compiler scheduling fence is needed, not a workgroup barrier.
__device__ __forceinline__ void wavebar() { __builtin_amdgcn_wave_barrier(); }

// softplus(x) for x in [-10, 10]: native v_exp_f32 / v_log_f32 path.
// exp(x) in [4.5e-5, 2.2e4], 1+e^x well-conditioned -> approx intrinsics safe.
__device__ __forceinline__ float softplus_clipped(float x) {
    return __logf(1.f + __expf(x));
}

__global__ __launch_bounds__(WAVES * 32) void cbow_main(
    const int*   __restrict__ tgt,      // [B]
    const int*   __restrict__ ctx,      // [B, C]
    const int*   __restrict__ neg,      // [B, N]
    const float* __restrict__ ctab,     // [V, 128]
    const float* __restrict__ otab,     // [V, 128]
    float*       __restrict__ partial)  // [gridDim.x]
{
    __shared__ _Float16 sh[WAVES][2][TILE * PITCH]; // [wave][A/B][tile]
    __shared__ float    wsum[WAVES];

    const int lane = threadIdx.x & 31;
    const int wave = threadIdx.x >> 5;
    const int tile = blockIdx.x * WAVES + wave;     // 0 .. B/16-1 (exact)
    const int b0   = tile * TILE;

    _Float16* Ab = &sh[wave][0][0];
    _Float16* Bb = &sh[wave][1][0];

    const int r  = lane & 15;   // row/col index inside tile
    const int hi = lane >> 4;   // half-wave select

    // ---- 1) gather-sum context rows -> mc tile in LDS as f16 ----
    for (int bi = 0; bi < TILE; ++bi) {
        const int b = b0 + bi;
        float ax = 0.f, ay = 0.f, az = 0.f, aw = 0.f;
        #pragma unroll
        for (int c = 0; c < CWIN; ++c) {
            const int row = ctx[b * CWIN + c];
            const float4 v = ((const float4*)(ctab + (size_t)row * EMB))[lane];
            ax += v.x; ay += v.y; az += v.z; aw += v.w;
        }
        h2* d2 = (h2*)(Ab + bi * PITCH + lane * 4);   // 8B-aligned
        h2 t0; t0.x = (_Float16)ax; t0.y = (_Float16)ay;
        h2 t1; t1.x = (_Float16)az; t1.y = (_Float16)aw;
        d2[0] = t0; d2[1] = t1;
    }
    wavebar();

    // ---- 2) hoist A fragments (16-bit A 16x32 layout, 4 K-steps) ----
    // lane m<16: VGPR0..3 = K 0..7, VGPR4..7 = K 16..23 ; lanes 16..31: +8
    v16h afr[4];
    #pragma unroll
    for (int kk = 0; kk < 4; ++kk) {
        const _Float16* base = Ab + r * PITCH + kk * 32 + 8 * hi;
        #pragma unroll
        for (int p = 0; p < 8; ++p) {
            const int K = (p < 4) ? 2 * p : 2 * p + 8;
            const h2 t = *(const h2*)(base + K);
            afr[kk][2 * p]     = t.x;
            afr[kk][2 * p + 1] = t.y;
        }
    }

    // software prefetch of chain 0 rows (global_prefetch_b8; lane*16B -> 512B/row)
    #pragma unroll
    for (int bi = 0; bi < TILE; ++bi) {
        const int row = tgt[b0 + bi];
        __builtin_prefetch(otab + (size_t)row * EMB + lane * 4, 0, 3);
    }

    // ---- 3) 21 score chains: j==0 target, j>=1 negative j-1 ----
    float lacc = 0.f;
    for (int j = 0; j < 1 + NNEG; ++j) {
        // prefetch next chain's rows while this chain is gathered/consumed
        if (j < NNEG) {
            #pragma unroll
            for (int bi = 0; bi < TILE; ++bi) {
                const int row = neg[(b0 + bi) * NNEG + j];
                __builtin_prefetch(otab + (size_t)row * EMB + lane * 4, 0, 3);
            }
        }
        // gather 16 output_table rows -> B tile (f16); unrolled => 16 loads in flight
        #pragma unroll
        for (int bi = 0; bi < TILE; ++bi) {
            const int b   = b0 + bi;
            const int row = (j == 0) ? tgt[b] : neg[b * NNEG + (j - 1)];
            const float4 v = ((const float4*)(otab + (size_t)row * EMB))[lane];
            h2* d2 = (h2*)(Bb + bi * PITCH + lane * 4);
            h2 t0; t0.x = (_Float16)v.x; t0.y = (_Float16)v.y;
            h2 t1; t1.x = (_Float16)v.z; t1.y = (_Float16)v.w;
            d2[0] = t0; d2[1] = t1;
        }
        wavebar();

        // B fragments: lane n<16: K 0..7, 16..23 ; lanes 16..31: +8 (same as A)
        v8f c = {};
        #pragma unroll
        for (int kk = 0; kk < 4; ++kk) {
            v16h bfr;
            const _Float16* base = Bb + r * PITCH + kk * 32 + 8 * hi;
            #pragma unroll
            for (int p = 0; p < 8; ++p) {
                const h2 t = *(const h2*)(base + ((p < 4) ? 2 * p : 2 * p + 8));
                bfr[2 * p]     = t.x;
                bfr[2 * p + 1] = t.y;
            }
            c = __builtin_amdgcn_wmma_f32_16x16x32_f16(
                    false, afr[kk], false, bfr, (short)0, c, false, false);
        }
        wavebar();

        // diagonal: lane l<8 -> D[l,l] in c[l]; lane>=24 -> D[l-16,l-16] in c[l-24]
        // branch-free log-depth select (7 v_cndmask), then masked accumulate
        const int m = lane & 7;
        const float s01 = (m & 1) ? c[1] : c[0];
        const float s23 = (m & 1) ? c[3] : c[2];
        const float s45 = (m & 1) ? c[5] : c[4];
        const float s67 = (m & 1) ? c[7] : c[6];
        const float sa  = (m & 2) ? s23 : s01;
        const float sb  = (m & 2) ? s67 : s45;
        float s         = (m & 4) ? sb  : sa;

        s = fminf(10.f, fmaxf(-10.f, s));
        const float x    = (j == 0) ? -s : s;          // softplus argument
        const float loss = softplus_clipped(x);        // native exp/log, |x|<=10
        const bool  act  = (lane < 8) || (lane >= 24); // 16 diag owners
        lacc += act ? loss : 0.f;
    }

    // ---- 4) wave + block reduction (fixed order -> deterministic) ----
    #pragma unroll
    for (int off = 16; off; off >>= 1) lacc += __shfl_xor(lacc, off, 32);
    if (lane == 0) wsum[wave] = lacc;
    __syncthreads();
    if (threadIdx.x == 0)
        partial[blockIdx.x] = (wsum[0] + wsum[1]) + (wsum[2] + wsum[3]);
}

__global__ __launch_bounds__(256) void cbow_reduce(
    const float* __restrict__ partial, float* __restrict__ out,
    int n, float scale)
{
    __shared__ float s[256];
    float v = 0.f;
    for (int i = threadIdx.x; i < n; i += 256) v += partial[i];
    s[threadIdx.x] = v;
    __syncthreads();
    #pragma unroll
    for (int off = 128; off; off >>= 1) {
        if ((int)threadIdx.x < off) s[threadIdx.x] += s[threadIdx.x + off];
        __syncthreads();
    }
    if (threadIdx.x == 0) out[0] = s[0] * scale;
}

extern "C" void kernel_launch(void* const* d_in, const int* in_sizes, int n_in,
                              void* d_out, int out_size, void* d_ws, size_t ws_size,
                              hipStream_t stream) {
    const int*   tgt  = (const int*)d_in[0];    // pos_target   [B]
    const int*   ctx  = (const int*)d_in[1];    // pos_contexts [B,C]
    const int*   neg  = (const int*)d_in[2];    // pos_negatives[B,N]
    const float* ctab = (const float*)d_in[3];  // context_table[V,128]
    const float* otab = (const float*)d_in[4];  // output_table [V,128]
    float* partial = (float*)d_ws;              // 256 floats of scratch
    float* out     = (float*)d_out;

    const int nblk = BTOT / (TILE * WAVES);     // 256
    cbow_main<<<nblk, WAVES * 32, 0, stream>>>(tgt, ctx, neg, ctab, otab, partial);
    cbow_reduce<<<1, 256, 0, stream>>>(partial, out, nblk, 1.0f / (float)BTOT);
}
